// PRET_24833500905908
// MI455X (gfx1250) — compile-verified, matrix operand order checked by reading, not run on hardware
//
#include <hip/hip_runtime.h>
#include <hip/hip_bf16.h>
#include <math.h>

// ---- CDNA5 (gfx1250) fused BERT self-attention block, bf16 WMMA + f32 softmax/LN ----

typedef __bf16 bf16;
typedef __attribute__((ext_vector_type(16))) __bf16 v16bf;
typedef __attribute__((ext_vector_type(8)))  float  v8f;
typedef unsigned int u32x4 __attribute__((ext_vector_type(4)));

#define D_MODEL 768
#define NHEAD   12
#define DHEAD   64
#define BATCH   8
#define SEQ     1024
#define ROWS    (BATCH * SEQ)   // 8192

// A/B fragment for v_wmma_f32_16x16x32_bf16:
// lane l holds row/col (l&15); 8 contiguous bf16 at K = kb + 8*(l>>4),
// 8 more at K = kb + 16 + 8*(l>>4). Two 16-byte loads.
__device__ __forceinline__ v16bf load_frag16(const bf16* p) {
  union { v16bf v; u32x4 u[2]; } f;
  f.u[0] = *(const u32x4*)(p);
  f.u[1] = *(const u32x4*)(p + 16);
  return f.v;
}

__device__ __forceinline__ v8f wmma_bf16(v16bf a, v16bf b, v8f c) {
  return __builtin_amdgcn_wmma_f32_16x16x32_bf16(false, a, false, b, (short)0, c, false, false);
}

// ---------------- kernel 1: f32 -> bf16 convert ----------------
__global__ void k_cvt_bf16(const float* __restrict__ src, bf16* __restrict__ dst, int n) {
  int i = blockIdx.x * blockDim.x + threadIdx.x;
  int stride = gridDim.x * blockDim.x;
  for (; i < n; i += stride) dst[i] = (bf16)src[i];
}

// ---------------- kernel 2: weight transpose (k,n)->(n,k) + bf16 ----------------
__global__ void k_transpose_w(const float* __restrict__ W, bf16* __restrict__ Wt) {
  int t = blockIdx.x * 256 + threadIdx.x;
  if (t >= D_MODEL * D_MODEL) return;
  int k = t / D_MODEL, n = t % D_MODEL;
  Wt[(size_t)n * D_MODEL + k] = (bf16)W[(size_t)k * D_MODEL + n];
}

// ---------------- kernel 3: QKV projection GEMM ----------------
// C[8192 x 768] = Xb @ W + bias, written bf16 in (B,H,L,DH) head layout.
// One wave per 32x32 tile (2x2 wmma register blocking). grid = 768 blocks x 256 thr.
__global__ void __launch_bounds__(256) k_gemm_qkv(const bf16* __restrict__ Xb,
                                                  const bf16* __restrict__ Wt,
                                                  const float* __restrict__ bias,
                                                  bf16* __restrict__ out) {
  const int lane = threadIdx.x & 31;
  const int wave = threadIdx.x >> 5;
  const int g  = blockIdx.x * 8 + wave;      // 6144 waves total
  const int mi = g / 24, ni = g % 24;        // 256 x 24 tiles of 32x32
  const int Mb = mi * 32, Nb = ni * 32;
  const int l15 = lane & 15, lh = lane >> 4;

  v8f acc[2][2] = {};
  const bf16* a0p = Xb + (size_t)(Mb + l15) * D_MODEL + 8 * lh;
  const bf16* a1p = a0p + (size_t)16 * D_MODEL;
  const bf16* b0p = Wt + (size_t)(Nb + l15) * D_MODEL + 8 * lh;
  const bf16* b1p = b0p + (size_t)16 * D_MODEL;

  for (int kb = 0; kb < D_MODEL; kb += 32) {
    v16bf a0 = load_frag16(a0p + kb);
    v16bf a1 = load_frag16(a1p + kb);
    v16bf b0 = load_frag16(b0p + kb);
    v16bf b1 = load_frag16(b1p + kb);
    acc[0][0] = wmma_bf16(a0, b0, acc[0][0]);
    acc[0][1] = wmma_bf16(a0, b1, acc[0][1]);
    acc[1][0] = wmma_bf16(a1, b0, acc[1][0]);
    acc[1][1] = wmma_bf16(a1, b1, acc[1][1]);
  }

#pragma unroll
  for (int i = 0; i < 2; ++i) {
#pragma unroll
    for (int j = 0; j < 2; ++j) {
      int N = Nb + j * 16 + l15;
      float bv = bias[N];
      int h = N >> 6, d = N & 63;
#pragma unroll
      for (int r = 0; r < 8; ++r) {
        int M = Mb + i * 16 + r + 8 * lh;         // C layout: M = r + 8*(lane/16)
        int bidx = M >> 10, pos = M & 1023;
        float v = acc[i][j][r] + bv;
        out[(((size_t)(bidx * NHEAD + h)) * SEQ + pos) * DHEAD + d] = (bf16)v;
      }
    }
  }
}

// ---------------- kernel 4: flash attention ----------------
// grid = 96 (b,h) * 8 q-groups = 768 blocks, 256 thr (8 waves, 16 q-rows each).
__global__ void __launch_bounds__(256) k_attn(const bf16* __restrict__ Q,
                                              const bf16* __restrict__ K,
                                              const bf16* __restrict__ V,
                                              const float* __restrict__ mask,
                                              bf16* __restrict__ ctx) {
  __shared__ __align__(16) bf16 Vt[DHEAD][32];        // V chunk transposed: [d][key]
  __shared__ __align__(16) bf16 Plds[8][16][32];      // per-wave P staging (C->A reshape)

  const int tid = threadIdx.x;
  const int lane = tid & 31, wave = tid >> 5;
  const int l15 = lane & 15, lh = lane >> 4;
  const int qg = blockIdx.x & 7;
  const int bh = blockIdx.x >> 3;
  const int b = bh / NHEAD;
  const int h = bh % NHEAD;
  const size_t base = (size_t)bh * SEQ * DHEAD;
  const int qBase = qg * 128 + wave * 16;

  const bf16* qrow = Q + base + (size_t)(qBase + l15) * DHEAD + 8 * lh;
  v16bf qf0 = load_frag16(qrow);        // d = 0..31
  v16bf qf1 = load_frag16(qrow + 32);   // d = 32..63

  v8f o0 = {}, o1 = {}, o2 = {}, o3 = {};
  float mrow[8], srow[8];
#pragma unroll
  for (int r = 0; r < 8; ++r) { mrow[r] = -1e30f; srow[r] = 0.0f; }
  const float* mptr = mask + (size_t)b * SEQ;

  for (int kc = 0; kc < SEQ; kc += 32) {
    __syncthreads();
    { // cooperative stage of V chunk, transposed into LDS
      int key = tid >> 3;
      int db  = (tid & 7) * 8;
      union { u32x4 u; bf16 hh[8]; } tmp;
      tmp.u = *(const u32x4*)(V + base + (size_t)(kc + key) * DHEAD + db);
#pragma unroll
      for (int j = 0; j < 8; ++j) Vt[db + j][key] = tmp.hh[j];
    }
    __syncthreads();

    // S = Q @ K^T for 32 keys (B frags contiguous: row = key, contraction = d)
    const bf16* k0p = K + base + (size_t)(kc + l15) * DHEAD + 8 * lh;
    const bf16* k1p = k0p + (size_t)16 * DHEAD;
    v16bf kf00 = load_frag16(k0p);
    v16bf kf01 = load_frag16(k0p + 32);
    v16bf kf10 = load_frag16(k1p);
    v16bf kf11 = load_frag16(k1p + 32);
    v8f s0 = {}, s1 = {};
    s0 = wmma_bf16(qf0, kf00, s0);
    s0 = wmma_bf16(qf1, kf01, s0);
    s1 = wmma_bf16(qf0, kf10, s1);
    s1 = wmma_bf16(qf1, kf11, s1);

    float mk0 = mptr[kc + l15];
    float mk1 = mptr[kc + 16 + l15];

#pragma unroll
    for (int r = 0; r < 8; ++r) {
      float a  = s0[r] * 0.125f + mk0;   // 1/sqrt(64)
      float bb = s1[r] * 0.125f + mk1;
      float mx = fmaxf(a, bb);
      mx = fmaxf(mx, __shfl_xor(mx, 1, 16));
      mx = fmaxf(mx, __shfl_xor(mx, 2, 16));
      mx = fmaxf(mx, __shfl_xor(mx, 4, 16));
      mx = fmaxf(mx, __shfl_xor(mx, 8, 16));
      float mnew = fmaxf(mrow[r], mx);
      float corr = __expf(mrow[r] - mnew);
      float p0 = __expf(a - mnew);
      float p1 = __expf(bb - mnew);
      float rs = p0 + p1;
      rs += __shfl_xor(rs, 1, 16);
      rs += __shfl_xor(rs, 2, 16);
      rs += __shfl_xor(rs, 4, 16);
      rs += __shfl_xor(rs, 8, 16);
      srow[r] = srow[r] * corr + rs;
      mrow[r] = mnew;
      o0[r] *= corr; o1[r] *= corr; o2[r] *= corr; o3[r] *= corr;
      int M = r + 8 * lh;
      Plds[wave][M][l15]      = (bf16)p0;   // C-layout -> LDS (in-order DS per wave)
      Plds[wave][M][16 + l15] = (bf16)p1;
    }

    // re-read P in A-fragment layout, V in B-fragment layout from LDS
    v16bf pa  = load_frag16(&Plds[wave][l15][8 * lh]);
    v16bf vb0 = load_frag16(&Vt[ 0 + l15][8 * lh]);
    v16bf vb1 = load_frag16(&Vt[16 + l15][8 * lh]);
    v16bf vb2 = load_frag16(&Vt[32 + l15][8 * lh]);
    v16bf vb3 = load_frag16(&Vt[48 + l15][8 * lh]);
    o0 = wmma_bf16(pa, vb0, o0);
    o1 = wmma_bf16(pa, vb1, o1);
    o2 = wmma_bf16(pa, vb2, o2);
    o3 = wmma_bf16(pa, vb3, o3);
  }

#pragma unroll
  for (int r = 0; r < 8; ++r) {
    float inv = 1.0f / srow[r];
    int M = qBase + r + 8 * lh;
    size_t rowo = ((size_t)(b * SEQ) + M) * D_MODEL + h * DHEAD;
    ctx[rowo +  0 + l15] = (bf16)(o0[r] * inv);
    ctx[rowo + 16 + l15] = (bf16)(o1[r] * inv);
    ctx[rowo + 32 + l15] = (bf16)(o2[r] * inv);
    ctx[rowo + 48 + l15] = (bf16)(o3[r] * inv);
  }
}

// ---------------- kernel 5: output GEMM + residual + LayerNorm ----------------
// grid = 512 blocks (16 rows each), 256 thr; wave w owns 96 output columns.
__global__ void __launch_bounds__(256) k_out_ln(const bf16* __restrict__ ctx,
                                                const bf16* __restrict__ Wot,
                                                const float* __restrict__ bo,
                                                const float* __restrict__ X,
                                                const float* __restrict__ gamma,
                                                const float* __restrict__ beta,
                                                float* __restrict__ out) {
  __shared__ float rowbuf[16][D_MODEL];   // 48 KB
  const int tid = threadIdx.x, lane = tid & 31, wave = tid >> 5;
  const int l15 = lane & 15, lh = lane >> 4;
  const int Mb = blockIdx.x * 16;
  const int Nw = wave * 96;

  v8f acc[6] = {};
  const bf16* aptr = ctx + (size_t)(Mb + l15) * D_MODEL + 8 * lh;
  const bf16* bptr[6];
#pragma unroll
  for (int t = 0; t < 6; ++t)
    bptr[t] = Wot + (size_t)(Nw + t * 16 + l15) * D_MODEL + 8 * lh;

  for (int kb = 0; kb < D_MODEL; kb += 32) {
    v16bf a = load_frag16(aptr + kb);
#pragma unroll
    for (int t = 0; t < 6; ++t) {
      v16bf bfr = load_frag16(bptr[t] + kb);
      acc[t] = wmma_bf16(a, bfr, acc[t]);
    }
  }

#pragma unroll
  for (int t = 0; t < 6; ++t) {
    int N = Nw + t * 16 + l15;
    float bv = bo[N];
#pragma unroll
    for (int r = 0; r < 8; ++r) {
      int M = r + 8 * lh;
      rowbuf[M][N] = acc[t][r] + bv + X[(size_t)(Mb + M) * D_MODEL + N];
    }
  }
  __syncthreads();

  int row = tid >> 4;       // 16 threads per row (lane groups of 16)
  int c0  = tid & 15;
  float sum = 0.0f, sq = 0.0f;
#pragma unroll 8
  for (int j = 0; j < 48; ++j) {
    float x = rowbuf[row][c0 + j * 16];
    sum += x; sq += x * x;
  }
  sum += __shfl_xor(sum, 1, 16); sq += __shfl_xor(sq, 1, 16);
  sum += __shfl_xor(sum, 2, 16); sq += __shfl_xor(sq, 2, 16);
  sum += __shfl_xor(sum, 4, 16); sq += __shfl_xor(sq, 4, 16);
  sum += __shfl_xor(sum, 8, 16); sq += __shfl_xor(sq, 8, 16);
  float mean = sum * (1.0f / D_MODEL);
  float var  = sq * (1.0f / D_MODEL) - mean * mean;
  float rstd = rsqrtf(fmaxf(var, 0.0f) + 1e-12f);
#pragma unroll 8
  for (int j = 0; j < 48; ++j) {
    int c = c0 + j * 16;
    float x = (rowbuf[row][c] - mean) * rstd;
    out[(size_t)(Mb + row) * D_MODEL + c] = x * gamma[c] + beta[c];
  }
}

// ---------------- host side ----------------
extern "C" void kernel_launch(void* const* d_in, const int* in_sizes, int n_in,
                              void* d_out, int out_size, void* d_ws, size_t ws_size,
                              hipStream_t stream) {
  const float* hidden = (const float*)d_in[0];
  const float* amask  = (const float*)d_in[1];
  const float* Wq = (const float*)d_in[2];
  const float* bq = (const float*)d_in[3];
  const float* Wk = (const float*)d_in[4];
  const float* bk = (const float*)d_in[5];
  const float* Wv = (const float*)d_in[6];
  const float* bv = (const float*)d_in[7];
  const float* Wo = (const float*)d_in[8];
  const float* bo = (const float*)d_in[9];
  const float* gamma = (const float*)d_in[10];
  const float* beta  = (const float*)d_in[11];
  float* out = (float*)d_out;

  char* ws = (char*)d_ws;
  size_t off = 0;
  const size_t SX = (size_t)ROWS * D_MODEL * sizeof(bf16);       // 12 MB
  const size_t SW = (size_t)D_MODEL * D_MODEL * sizeof(bf16);    // 1.125 MB
  bf16* Xb  = (bf16*)(ws + off); off += SX;
  bf16* Wqt = (bf16*)(ws + off); off += SW;
  bf16* Wkt = (bf16*)(ws + off); off += SW;
  bf16* Wvt = (bf16*)(ws + off); off += SW;
  bf16* Wot = (bf16*)(ws + off); off += SW;
  bf16* Qb  = (bf16*)(ws + off); off += SX;
  bf16* Kb  = (bf16*)(ws + off); off += SX;
  bf16* Vb  = (bf16*)(ws + off); off += SX;
  bf16* Ctx = (bf16*)(ws + off); off += SX;

  k_cvt_bf16<<<512, 256, 0, stream>>>(hidden, Xb, ROWS * D_MODEL);
  const int wn = D_MODEL * D_MODEL;
  k_transpose_w<<<(wn + 255) / 256, 256, 0, stream>>>(Wq, Wqt);
  k_transpose_w<<<(wn + 255) / 256, 256, 0, stream>>>(Wk, Wkt);
  k_transpose_w<<<(wn + 255) / 256, 256, 0, stream>>>(Wv, Wvt);
  k_transpose_w<<<(wn + 255) / 256, 256, 0, stream>>>(Wo, Wot);

  k_gemm_qkv<<<768, 256, 0, stream>>>(Xb, Wqt, bq, Qb);
  k_gemm_qkv<<<768, 256, 0, stream>>>(Xb, Wkt, bk, Kb);
  k_gemm_qkv<<<768, 256, 0, stream>>>(Xb, Wvt, bv, Vb);

  k_attn<<<768, 256, 0, stream>>>(Qb, Kb, Vb, amask, Ctx);

  k_out_ln<<<512, 256, 0, stream>>>(Ctx, Wot, bo, hidden, gamma, beta, out);
}
